// VQ_VAE_z32_7859790152058
// MI455X (gfx1250) — matrix operand order repaired and checked by reading, not compiled, not run on hardware
//
#include <hip/hip_runtime.h>
#include <hip/hip_bf16.h>

typedef __bf16 bf16_t;
typedef __attribute__((ext_vector_type(16))) __bf16 v16bf;
typedef __attribute__((ext_vector_type(8)))  float  v8f;

// ---------------------------------------------------------------------------
// Elementwise helpers
// ---------------------------------------------------------------------------
__global__ void k_zero_f32(float* p, int n) {
    int i = blockIdx.x * blockDim.x + threadIdx.x;
    if (i < n) p[i] = 0.f;
}

__global__ void k_f32_to_bf16(const float* __restrict__ x, bf16_t* __restrict__ y, int n) {
    int i = blockIdx.x * blockDim.x + threadIdx.x;
    if (i < n) y[i] = (bf16_t)x[i];
}

__global__ void k_relu_f32_to_bf16(const float* __restrict__ x, bf16_t* __restrict__ y, int n) {
    int i = blockIdx.x * blockDim.x + threadIdx.x;
    if (i < n) y[i] = (bf16_t)fmaxf(x[i], 0.f);
}

__global__ void k_add_inplace(float* __restrict__ x, const float* __restrict__ h, int n) {
    int i = blockIdx.x * blockDim.x + threadIdx.x;
    if (i < n) x[i] += h[i];
}

// ---------------------------------------------------------------------------
// Weight repack: f32 conv weights -> bf16 GEMM-B laid out as
//   Bp[kchunk][Cout][32]   (kchunk = k/32, inner = k%32)
// so that one lane's 16 chunk-values are 32 contiguous, 32B-aligned bytes.
// K ordered as (c, dr, dc); kh = khB + khS*dr, kw = kwB + kwS*dc.
// ---------------------------------------------------------------------------
__global__ void k_repack(const float* __restrict__ W, bf16_t* __restrict__ Bp,
                         int O, int TH, int TW,
                         int khB, int khS, int kwB, int kwS,
                         int wsO, int wsC, int wsH, int wsW, int total) {
    int i = blockIdx.x * blockDim.x + threadIdx.x;
    if (i >= total) return;
    int k = i / O;
    int o = i - k * O;
    int TT = TH * TW;
    int c  = k / TT;
    int r  = k - c * TT;
    int dr = r / TW;
    int dc = r - dr * TW;
    int kh = khB + khS * dr;
    int kw = kwB + kwS * dc;
    float w = W[(size_t)o * wsO + (size_t)c * wsC + kh * wsH + kw * wsW];
    Bp[((size_t)(k >> 5) * O + o) * 32 + (k & 31)] = (bf16_t)w;
}

// ---------------------------------------------------------------------------
// Unified implicit-GEMM conv on the WMMA bf16 tensor path.
// 2D grid: blockIdx.x*8 + waveInBlock = M-tile (16 spatial rows),
//          blockIdx.y = 64-column N-tile (4 accumulators, one shared A frag).
// 4x v_wmma_f32_16x16x32_bf16 per 32-deep K chunk; next B panel prefetched
// (global_prefetch_b8).  Template (TH,TW) => constant-divisor tap decompose;
// all spatial dims are powers of two and passed as shifts.
// Normal convs: om=1, stride=s, dy = dr - pad.
// Transposed-conv parity classes: om=2, stride=1, dy = parity - dr,
// scatter to (oh*2+oy0, ow*2+ox0).
// ---------------------------------------------------------------------------
template <int TH, int TW>
__global__ void __launch_bounds__(256)
k_conv_wmma(const bf16_t* __restrict__ X, const bf16_t* __restrict__ Bp,
            const float* __restrict__ bias, float* __restrict__ Y,
            int Cin, int hinSh, int winSh, int Cout,
            int hoSh, int woSh, int Hof, int Wof,
            int oy0, int ox0, int om, int stride,
            int dyB, int dyS, int dxB, int dxS,
            int K, int M) {
    constexpr int TT = TH * TW;
    int mTile = blockIdx.x * 8 + (threadIdx.x >> 5);
    int nTile = blockIdx.y;
    int lane  = threadIdx.x & 31;
    if (mTile * 16 >= M) return;

    const int Hin = 1 << hinSh, Win = 1 << winSh;
    const int Wo  = 1 << woSh;

    int half = lane >> 4;   // which K-half this lane carries (ISA 7.12.2)
    int l16  = lane & 15;

    // A operand: this lane holds row m = mTile*16 + l16
    int mA = mTile * 16 + l16;
    bool mAok = (mA < M);
    int owA  = mA & (Wo - 1);
    int tA   = mA >> woSh;
    int ohA  = tA & ((1 << hoSh) - 1);
    int imgA = tA >> hoSh;
    int baseY = ohA * stride + dyB;
    int baseX = owA * stride + dxB;
    const bf16_t* Ximg = X + ((size_t)imgA * Cin << (hinSh + winSh));

    int n0  = nTile * 64 + l16;           // first of 4 column subtiles
    int kbA = half ? 16 : 0;              // A: lanes 0-15 K 0-15; lanes 16-31 K 16-31
    int kbB = half ? 16 : 0;              // B: lanes 0-15 K 0-15; lanes 16-31 K 16-31

    v16bf bzero;
#pragma unroll
    for (int j = 0; j < 16; ++j) bzero[j] = (bf16_t)0.f;

    v8f acc[4];
#pragma unroll
    for (int s = 0; s < 4; ++s)
#pragma unroll
        for (int i = 0; i < 8; ++i) acc[s][i] = 0.f;

#pragma unroll 2
    for (int k0 = 0; k0 < K; k0 += 32) {
        v16bf a;
#pragma unroll
        for (int j = 0; j < 16; ++j) {
            int kk = k0 + kbA + j;
            bf16_t av = (bf16_t)0.f;
            if (mAok) {
                int c  = kk / TT;           // TT is a compile-time constant
                int r  = kk - c * TT;
                int dr = r / TW;
                int dc = r - dr * TW;
                int ih = baseY + dyS * dr;
                int iw = baseX + dxS * dc;
                if ((unsigned)ih < (unsigned)Hin && (unsigned)iw < (unsigned)Win)
                    av = Ximg[((((size_t)c << hinSh) + ih) << winSh) + iw];
            }
            a[j] = av;
        }
        const bf16_t* brow = Bp + ((size_t)(k0 >> 5) * Cout) * 32 + kbB;
        // prefetch next K chunk's B panel into cache (global_prefetch_b8)
        if (k0 + 32 < K)
            __builtin_prefetch(brow + (size_t)Cout * 32 + (size_t)n0 * 32, 0, 3);
#pragma unroll
        for (int s = 0; s < 4; ++s) {
            int n = n0 + 16 * s;
            v16bf b = (n < Cout) ? *(const v16bf*)(brow + (size_t)n * 32) : bzero;
            acc[s] = __builtin_amdgcn_wmma_f32_16x16x32_bf16(false, a, false, b,
                                                             (short)0, acc[s], false, false);
        }
    }

    // D layout: VGPR i -> M = mTile*16 + (half ? i+8 : i), N = l16 (+16s)
    float bv[4];
#pragma unroll
    for (int s = 0; s < 4; ++s) {
        int n = n0 + 16 * s;
        bv[s] = (n < Cout) ? bias[n] : 0.f;
    }
    size_t chanStride = (size_t)Hof * Wof * 16;
#pragma unroll
    for (int i = 0; i < 8; ++i) {
        int mD = mTile * 16 + (half ? (i + 8) : i);
        if (mD >= M) continue;
        int ow  = mD & (Wo - 1);
        int t   = mD >> woSh;
        int oh  = t & ((1 << hoSh) - 1);
        int img = t >> hoSh;
        size_t o0 = (((size_t)img * Cout + n0) * Hof + (oh * om + oy0)) * Wof + (ow * om + ox0);
#pragma unroll
        for (int s = 0; s < 4; ++s) {
            int n = n0 + 16 * s;
            if (n < Cout) Y[o0 + (size_t)s * chanStride] = acc[s][i] + bv[s];
        }
    }
}

// ---------------------------------------------------------------------------
// BatchNorm training-mode batch statistics (mean & biased var per channel).
// ---------------------------------------------------------------------------
__global__ void k_bn_stats(const float* __restrict__ X, float* __restrict__ mean,
                           float* __restrict__ var, int N, int C, int HW) {
    int c = blockIdx.x;
    __shared__ float s1[256], s2[256];
    float sum = 0.f, sq = 0.f;
    for (int n = 0; n < N; ++n) {
        const float* p = X + ((size_t)n * C + c) * HW;
        for (int i = threadIdx.x; i < HW; i += blockDim.x) {
            float v = p[i];
            sum += v;
            sq  += v * v;
        }
    }
    s1[threadIdx.x] = sum;
    s2[threadIdx.x] = sq;
    __syncthreads();
    for (int off = 128; off > 0; off >>= 1) {
        if ((int)threadIdx.x < off) {
            s1[threadIdx.x] += s1[threadIdx.x + off];
            s2[threadIdx.x] += s2[threadIdx.x + off];
        }
        __syncthreads();
    }
    if (threadIdx.x == 0) {
        float cnt = (float)N * (float)HW;
        float m = s1[0] / cnt;
        mean[c] = m;
        var[c]  = s2[0] / cnt - m * m;
    }
}

// Apply BN (+optional ReLU).  HW and C are powers of two: c = (i>>hwSh)&cMask.
__global__ void k_bn_apply(const float* __restrict__ X, const float* __restrict__ mean,
                           const float* __restrict__ var, const float* __restrict__ g,
                           const float* __restrict__ b, float* __restrict__ Yf,
                           bf16_t* __restrict__ Yb, int hwSh, int cMask, int n, int relu) {
    int i = blockIdx.x * blockDim.x + threadIdx.x;
    if (i >= n) return;
    int c = (i >> hwSh) & cMask;
    float v = g[c] * (X[i] - mean[c]) * rsqrtf(var[c] + 1e-5f) + b[c];
    if (relu) v = fmaxf(v, 0.f);
    if (Yf) Yf[i] = v;
    if (Yb) Yb[i] = (bf16_t)v;
}

// ---------------------------------------------------------------------------
// Vector quantizer
// ---------------------------------------------------------------------------
__global__ void k_code_norms(const float* __restrict__ E, float* __restrict__ enorm) {
    int n = blockIdx.x * blockDim.x + threadIdx.x;
    if (n < 512) {
        float s = 0.f;
        const float* e = E + (size_t)n * 128;
        for (int d = 0; d < 128; ++d) s += e[d] * e[d];
        enorm[n] = s;
    }
}

// One wave per spatial vector m (65536).  z-row staged in LDS; lanes scan
// codes n = lane, lane+32, ...; jnp.argmin tie-breaking (lowest index);
// wave32 shuffle min+index reduction.
__global__ void k_vq_assign(const float* __restrict__ Z, const float* __restrict__ E,
                            const float* __restrict__ enorm, int* __restrict__ idx,
                            float* __restrict__ counts) {
    int wib  = threadIdx.x >> 5;
    int lane = threadIdx.x & 31;
    int m = blockIdx.x * 8 + wib;
    __shared__ float zs[8][128];
    int img = m >> 10;
    int rem = m & 1023;
    int oh  = rem >> 5;
    int ow  = rem & 31;
    const float* zp = Z + (((size_t)img * 128) * 32 + oh) * 32 + ow;
    for (int d = lane; d < 128; d += 32) zs[wib][d] = zp[(size_t)d * 1024];
    __syncthreads();

    float best = 3.402823e38f;
    int bestn = 0;
    for (int n = lane; n < 512; n += 32) {
        const float* e = E + (size_t)n * 128;
        float dot = 0.f;
        for (int d = 0; d < 128; ++d) dot += zs[wib][d] * e[d];
        float score = enorm[n] - 2.f * dot;
        if (score < best) { best = score; bestn = n; }
    }
    for (int off = 16; off > 0; off >>= 1) {
        float ov = __shfl_down(best, off, 32);
        int   oi = __shfl_down(bestn, off, 32);
        if (ov < best || (ov == best && oi < bestn)) { best = ov; bestn = oi; }
    }
    if (lane == 0) {
        idx[m] = bestn;
        atomicAdd(&counts[bestn], 1.0f);  // exact integer-valued: deterministic
    }
}

// zq = E[idx] in NCHW + per-block partial sums of (q - z)^2.
__global__ void k_zq_build(const float* __restrict__ E, const int* __restrict__ idx,
                           const float* __restrict__ Z, float* __restrict__ ZQ,
                           float* __restrict__ part) {
    int i = blockIdx.x * 256 + threadIdx.x;   // over 65536*128
    int m = i >> 7;
    int d = i & 127;
    float q = E[(size_t)idx[m] * 128 + d];
    int img = m >> 10;
    int rem = m & 1023;
    int oh  = rem >> 5;
    int ow  = rem & 31;
    size_t o = ((((size_t)img * 128 + d) * 32 + oh) * 32 + ow);
    float z = Z[o];
    ZQ[o] = q;
    float df = q - z;
    float v = df * df;
    __shared__ float s[256];
    s[threadIdx.x] = v;
    __syncthreads();
    for (int off = 128; off > 0; off >>= 1) {
        if ((int)threadIdx.x < off) s[threadIdx.x] += s[threadIdx.x + off];
        __syncthreads();
    }
    if (threadIdx.x == 0) part[blockIdx.x] = s[0];
}

__global__ void k_mse_partial(const float* __restrict__ A, const float* __restrict__ B,
                              float* __restrict__ part) {
    int i = blockIdx.x * 256 + threadIdx.x;
    float df = A[i] - B[i];
    float v = df * df;
    __shared__ float s[256];
    s[threadIdx.x] = v;
    __syncthreads();
    for (int off = 128; off > 0; off >>= 1) {
        if ((int)threadIdx.x < off) s[threadIdx.x] += s[threadIdx.x + off];
        __syncthreads();
    }
    if (threadIdx.x == 0) part[blockIdx.x] = s[0];
}

__global__ void k_finalize(const float* __restrict__ partA, int nA,
                           const float* __restrict__ partB, int nB,
                           const float* __restrict__ counts, float* __restrict__ out) {
    __shared__ float sA[256], sB[256], sH[256];
    float a = 0.f, b = 0.f, h = 0.f;
    for (int i = threadIdx.x; i < nA; i += 256) a += partA[i];
    for (int i = threadIdx.x; i < nB; i += 256) b += partB[i];
    for (int i = threadIdx.x; i < 512; i += 256) {
        float p = counts[i] * (1.f / 65536.f);
        h -= p * logf(p + 1e-10f);
    }
    sA[threadIdx.x] = a; sB[threadIdx.x] = b; sH[threadIdx.x] = h;
    __syncthreads();
    for (int off = 128; off > 0; off >>= 1) {
        if ((int)threadIdx.x < off) {
            sA[threadIdx.x] += sA[threadIdx.x + off];
            sB[threadIdx.x] += sB[threadIdx.x + off];
            sH[threadIdx.x] += sH[threadIdx.x + off];
        }
        __syncthreads();
    }
    if (threadIdx.x == 0) {
        float recon = sB[0] / 2097152.f;
        float cl = 1.25f * (sA[0] / 8388608.f);   // q_latent + 0.25 * e_latent
        out[0] = recon;
        out[1] = cl;
        out[2] = expf(sH[0]);
        out[3] = recon + cl;
    }
}

// ---------------------------------------------------------------------------
// Host-side launch helpers
// ---------------------------------------------------------------------------
static void launch_repack(hipStream_t s, const float* W, bf16_t* Bm, int O, int Cin,
                          int TH, int TW, int khB, int khS, int kwB, int kwS,
                          int wsO, int wsC, int wsH, int wsW) {
    int K = Cin * TH * TW;
    int n = K * O;
    k_repack<<<(n + 255) / 256, 256, 0, s>>>(W, Bm, O, TH, TW, khB, khS, kwB, kwS,
                                             wsO, wsC, wsH, wsW, n);
}

static int ilog2(int x) { int s = 0; while ((1 << s) < x) ++s; return s; }

static void launch_conv(hipStream_t s, const bf16_t* X, const bf16_t* Bm,
                        const float* bias, float* Y,
                        int N, int Cin, int Hin, int Win,
                        int Cout, int Ho, int Wo, int Hof, int Wof,
                        int oy0, int ox0, int om, int stride,
                        int TH, int TW, int dyB, int dyS, int dxB, int dxS) {
    int K = Cin * TH * TW;
    int M = N * Ho * Wo;
    int mT = (M + 15) >> 4;
    int nT = (Cout + 63) >> 6;            // 64 columns per wave
    dim3 grid((mT + 7) / 8, nT, 1);       // 8 waves (256 threads) per block
    int hs = ilog2(Hin), ws = ilog2(Win), hos = ilog2(Ho), wos = ilog2(Wo);
#define CONV_ARGS X, Bm, bias, Y, Cin, hs, ws, Cout, hos, wos, Hof, Wof, \
                  oy0, ox0, om, stride, dyB, dyS, dxB, dxS, K, M
    if (TH == 4)      k_conv_wmma<4, 4><<<grid, 256, 0, s>>>(CONV_ARGS);
    else if (TH == 3) k_conv_wmma<3, 3><<<grid, 256, 0, s>>>(CONV_ARGS);
    else if (TH == 2) k_conv_wmma<2, 2><<<grid, 256, 0, s>>>(CONV_ARGS);
    else              k_conv_wmma<1, 1><<<grid, 256, 0, s>>>(CONV_ARGS);
#undef CONV_ARGS
}

// relu -> bn(conv3x3) -> relu -> bn(conv1x1) -> x += h, all on [64,128,32,32]
static void resblock(hipStream_t s, float* x,
                     const float* w1, const float* b1, const float* g1, const float* be1,
                     const float* w2, const float* b2, const float* g2, const float* be2,
                     bf16_t* hinbf, float* hmid, bf16_t* hmidbf, float* h2,
                     bf16_t* wB, float* mean, float* var) {
    const int nX = 64 * 128 * 32 * 32;   // 8388608
    const int nM = 64 * 256 * 32 * 32;   // 16777216
    k_relu_f32_to_bf16<<<nX / 256, 256, 0, s>>>(x, hinbf, nX);
    launch_repack(s, w1, wB, 256, 128, 3, 3, 0, 1, 0, 1, 128 * 9, 9, 3, 1);
    launch_conv(s, hinbf, wB, b1, hmid, 64, 128, 32, 32, 256, 32, 32, 32, 32,
                0, 0, 1, 1, 3, 3, -1, 1, -1, 1);
    k_bn_stats<<<256, 256, 0, s>>>(hmid, mean, var, 64, 256, 1024);
    k_bn_apply<<<nM / 256, 256, 0, s>>>(hmid, mean, var, g1, be1, nullptr, hmidbf,
                                        10, 255, nM, 1);
    launch_repack(s, w2, wB, 128, 256, 1, 1, 0, 1, 0, 1, 256, 1, 1, 1);
    launch_conv(s, hmidbf, wB, b2, h2, 64, 256, 32, 32, 128, 32, 32, 32, 32,
                0, 0, 1, 1, 1, 1, 0, 1, 0, 1);
    k_bn_stats<<<128, 256, 0, s>>>(h2, mean, var, 64, 128, 1024);
    k_bn_apply<<<nX / 256, 256, 0, s>>>(h2, mean, var, g2, be2, h2, nullptr,
                                        10, 127, nX, 0);
    k_add_inplace<<<nX / 256, 256, 0, s>>>(x, h2, nX);
}

// ---------------------------------------------------------------------------
extern "C" void kernel_launch(void* const* d_in, const int* in_sizes, int n_in,
                              void* d_out, int out_size, void* d_ws, size_t ws_size,
                              hipStream_t stream) {
    // ---- input mapping (setup_inputs dict order, params flattened) ----
    const float* inp     = (const float*)d_in[0];    // [64,2,128,128]
    const float* enc_w1  = (const float*)d_in[1];    // (64,2,4,4)
    const float* enc_b1  = (const float*)d_in[2];
    const float* enc_g1  = (const float*)d_in[3];
    const float* enc_be1 = (const float*)d_in[4];
    const float* enc_w2  = (const float*)d_in[5];    // (128,64,4,4)
    const float* enc_b2  = (const float*)d_in[6];
    const float* enc_g2  = (const float*)d_in[7];
    const float* enc_be2 = (const float*)d_in[8];
    const float* resE[2][8];
    for (int l = 0; l < 2; ++l)
        for (int j = 0; j < 8; ++j) resE[l][j] = (const float*)d_in[9 + l * 8 + j];
    const float* codebook = (const float*)d_in[25];  // (512,128)
    const float* resD[2][8];
    for (int l = 0; l < 2; ++l)
        for (int j = 0; j < 8; ++j) resD[l][j] = (const float*)d_in[26 + l * 8 + j];
    const float* dec_wt1 = (const float*)d_in[42];   // (128,64,4,4) torch IOHW
    const float* dec_bt1 = (const float*)d_in[43];
    const float* dec_g1  = (const float*)d_in[44];
    const float* dec_be1 = (const float*)d_in[45];
    const float* dec_wt2 = (const float*)d_in[46];   // (64,2,4,4) torch IOHW
    const float* dec_bt2 = (const float*)d_in[47];

    // ---- workspace carve (256B aligned) ----
    char* p = (char*)d_ws;
    auto carve = [&](size_t bytes) -> void* {
        void* r = (void*)p;
        p += (bytes + 255) & ~(size_t)255;
        return r;
    };
    bf16_t* xin_bf = (bf16_t*)carve(2097152 * sizeof(bf16_t));
    float*  z1     = (float*)carve(16777216 * sizeof(float));   // also reused as y1
    bf16_t* a1bf   = (bf16_t*)carve(16777216 * sizeof(bf16_t)); // also reused as y1bf
    float*  z2     = (float*)carve(8388608 * sizeof(float));
    float*  zb     = (float*)carve(8388608 * sizeof(float));    // encoder output z
    bf16_t* hinbf  = (bf16_t*)carve(8388608 * sizeof(bf16_t));
    float*  hmid   = (float*)carve(16777216 * sizeof(float));
    bf16_t* hmidbf = (bf16_t*)carve(16777216 * sizeof(bf16_t));
    float*  h2     = (float*)carve(8388608 * sizeof(float));
    float*  zq     = (float*)carve(8388608 * sizeof(float));
    bf16_t* wB     = (bf16_t*)carve(294912 * sizeof(bf16_t));   // largest B: 1152x256
    float*  mean   = (float*)carve(256 * sizeof(float));
    float*  var    = (float*)carve(256 * sizeof(float));
    float*  enorm  = (float*)carve(512 * sizeof(float));
    int*    idx    = (int*)carve(65536 * sizeof(int));
    float*  counts = (float*)carve(512 * sizeof(float));
    float*  partA  = (float*)carve(32768 * sizeof(float));
    float*  partB  = (float*)carve(8192 * sizeof(float));

    float* dec_out  = (float*)d_out;
    float* scal_out = dec_out + 2097152;

    // ---- init ----
    k_zero_f32<<<2, 256, 0, stream>>>(counts, 512);
    k_f32_to_bf16<<<2097152 / 256, 256, 0, stream>>>(inp, xin_bf, 2097152);

    // ---- encoder conv1: 2->64, k4 s2 p1, 128x128 -> 64x64 ----
    launch_repack(stream, enc_w1, wB, 64, 2, 4, 4, 0, 1, 0, 1, 32, 16, 4, 1);
    launch_conv(stream, xin_bf, wB, enc_b1, z1, 64, 2, 128, 128, 64, 64, 64, 64, 64,
                0, 0, 1, 2, 4, 4, -1, 1, -1, 1);
    k_bn_stats<<<64, 256, 0, stream>>>(z1, mean, var, 64, 64, 4096);
    k_bn_apply<<<16777216 / 256, 256, 0, stream>>>(z1, mean, var, enc_g1, enc_be1,
                                                   nullptr, a1bf, 12, 63, 16777216, 1);

    // ---- encoder conv2: 64->128, k4 s2 p1, 64x64 -> 32x32 ----
    launch_repack(stream, enc_w2, wB, 128, 64, 4, 4, 0, 1, 0, 1, 64 * 16, 16, 4, 1);
    launch_conv(stream, a1bf, wB, enc_b2, z2, 64, 64, 64, 64, 128, 32, 32, 32, 32,
                0, 0, 1, 2, 4, 4, -1, 1, -1, 1);
    k_bn_stats<<<128, 256, 0, stream>>>(z2, mean, var, 64, 128, 1024);
    k_bn_apply<<<8388608 / 256, 256, 0, stream>>>(z2, mean, var, enc_g2, enc_be2,
                                                  zb, nullptr, 10, 127, 8388608, 0);

    // ---- encoder residual stack ----
    for (int l = 0; l < 2; ++l)
        resblock(stream, zb, resE[l][0], resE[l][1], resE[l][2], resE[l][3],
                 resE[l][4], resE[l][5], resE[l][6], resE[l][7],
                 hinbf, hmid, hmidbf, h2, wB, mean, var);

    // ---- vector quantizer ----
    k_code_norms<<<2, 256, 0, stream>>>(codebook, enorm);
    k_vq_assign<<<8192, 256, 0, stream>>>(zb, codebook, enorm, idx, counts);
    k_zq_build<<<32768, 256, 0, stream>>>(codebook, idx, zb, zq, partA);

    // ---- decoder residual stack (on zq = q, straight-through forward) ----
    for (int l = 0; l < 2; ++l)
        resblock(stream, zq, resD[l][0], resD[l][1], resD[l][2], resD[l][3],
                 resD[l][4], resD[l][5], resD[l][6], resD[l][7],
                 hinbf, hmid, hmidbf, h2, wB, mean, var);

    // ---- convT1: 128->64, k4 s2 p1, 32x32 -> 64x64 (4 parity classes) ----
    k_f32_to_bf16<<<8388608 / 256, 256, 0, stream>>>(zq, hinbf, 8388608);
    for (int pr = 0; pr < 2; ++pr)
        for (int pc = 0; pc < 2; ++pc) {
            // weight IOHW (128,64,4,4): wsC = 64*16, wsO = 16; kh = (1-pr) + 2*dr
            launch_repack(stream, dec_wt1, wB, 64, 128, 2, 2,
                          1 - pr, 2, 1 - pc, 2, 16, 64 * 16, 4, 1);
            launch_conv(stream, hinbf, wB, dec_bt1, z1, 64, 128, 32, 32,
                        64, 32, 32, 64, 64, pr, pc, 2, 1, 2, 2, pr, -1, pc, -1);
        }
    k_bn_stats<<<64, 256, 0, stream>>>(z1, mean, var, 64, 64, 4096);
    k_bn_apply<<<16777216 / 256, 256, 0, stream>>>(z1, mean, var, dec_g1, dec_be1,
                                                   nullptr, a1bf, 12, 63, 16777216, 1);

    // ---- convT2: 64->2, k4 s2 p1, 64x64 -> 128x128, writes d_out directly ----
    for (int pr = 0; pr < 2; ++pr)
        for (int pc = 0; pc < 2; ++pc) {
            launch_repack(stream, dec_wt2, wB, 2, 64, 2, 2,
                          1 - pr, 2, 1 - pc, 2, 16, 2 * 16, 4, 1);
            launch_conv(stream, a1bf, wB, dec_bt2, dec_out, 64, 64, 64, 64,
                        2, 64, 64, 128, 128, pr, pc, 2, 1, 2, 2, pr, -1, pc, -1);
        }

    // ---- losses ----
    k_mse_partial<<<8192, 256, 0, stream>>>(dec_out, inp, partB);
    k_finalize<<<1, 256, 0, stream>>>(partA, 32768, partB, 8192, counts, scal_out);
}
